// SwinBlock_9045201125808
// MI455X (gfx1250) — matrix operand order repaired and checked by reading, not compiled, not run on hardware
//
#include <hip/hip_runtime.h>
#include <hip/hip_bf16.h>

// ---------------------------------------------------------------------------
// 3D Swin block for MI455X (gfx1250, wave32).  All GEMMs run on
// V_WMMA_F32_16X16X32_BF16 (bf16 inputs, f32 accumulate).
// Fragment layouts (A/B use this file's contiguous k order, not the ISA table order):
//   A (16x32 bf16): lane L: row = L%16, r = L/16, element e -> K = k0+16r+e
//   B (32x16 bf16): lane L: col = L%16, r = L/16, element e -> K = k0+16r+e
//   C/D (16x16 f32): lane L, vgpr j -> row = j + 8*(L/16), col = L%16
// QKV GEMM stages its B tile through LDS with GLOBAL_LOAD_ASYNC_TO_LDS_B128
// (ASYNCcnt-tracked, VGPR-bypassing) so 8 waves share one copy.
// ---------------------------------------------------------------------------

typedef __bf16 bf16;
typedef __attribute__((ext_vector_type(16))) __bf16 bf16x16;
typedef __attribute__((ext_vector_type(8)))  __bf16 bf16x8;
typedef __attribute__((ext_vector_type(8)))  float   f32x8;

#define DEVINL __device__ __forceinline__

// Problem constants
#define BATCH 2
#define DIMC 192
#define HEADS 6
#define HD 32
#define NWIN 512          // B * 256 windows
#define NTOK 98           // tokens per window
#define NPAD 112          // padded tokens (7*16)
#define KPAD 128          // padded attention K dim (4*32)
#define MROWS 50176       // NWIN * NTOK = B * L

DEVINL bf16x16 frag_cat(bf16x8 lo, bf16x8 hi) {
  bf16x16 r;
#pragma unroll
  for (int i = 0; i < 8; ++i) { r[i] = lo[i]; r[i + 8] = hi[i]; }
  return r;
}

// A-operand fragment from row-major [rows x ld] bf16 storage
DEVINL bf16x16 load_afrag(const bf16* p0, int row0, int ld, int k0) {
  int lane = threadIdx.x & 31;
  const bf16* p = p0 + (size_t)(row0 + (lane & 15)) * ld + k0 + ((lane >> 4) << 4);
  return frag_cat(*(const bf16x8*)p, *(const bf16x8*)(p + 8));
}

// B-operand fragment from row-major [N x K] storage (row index = output column n)
DEVINL bf16x16 load_bfrag(const bf16* p0, int n0, int ld, int k0) {
  int lane = threadIdx.x & 31;
  const bf16* p = p0 + (size_t)(n0 + (lane & 15)) * ld + k0 + ((lane >> 4) << 4);
  return frag_cat(*(const bf16x8*)p, *(const bf16x8*)(p + 8));
}

// Pre-packed B fragments: blob layout [kt][nt][lane][16 elems]
DEVINL bf16x16 load_bpack(const bf16* bp, int kt, int nt, int ntn) {
  int lane = threadIdx.x & 31;
  const bf16* p = bp + ((((size_t)kt * ntn + nt) * 32 + lane) << 4);
  return frag_cat(*(const bf16x8*)p, *(const bf16x8*)(p + 8));
}

// Packed B fragment from an LDS stage (blob index = kt*ntiles + nt_local)
DEVINL bf16x16 load_bpack_lds(const bf16* bs, int blob) {
  int lane = threadIdx.x & 31;
  const bf16* p = bs + (((blob * 32) + lane) << 4);
  return frag_cat(*(const bf16x8*)p, *(const bf16x8*)(p + 8));
}

DEVINL f32x8 wmma_bf16(bf16x16 a, bf16x16 b, f32x8 c) {
  return __builtin_amdgcn_wmma_f32_16x16x32_bf16(false, a, false, b, (short)0, c,
                                                 false, false);
}

// CDNA5 async global->LDS copy (16B per lane), tracked by ASYNCcnt.
// lds_off: LDS byte address (low 32 bits of the generic shared pointer).
DEVINL void async_copy_b128(unsigned lds_off, const void* gptr) {
  asm volatile("global_load_async_to_lds_b128 %0, %1, off"
               :: "v"(lds_off), "v"(gptr) : "memory");
}
DEVINL void wait_async0() {
  asm volatile("s_wait_asynccnt 0x0" ::: "memory");
}

// ---------------------------------------------------------------------------
// Pack a (N,K) row-major f32 weight into bf16 WMMA B-fragment blobs.
// ---------------------------------------------------------------------------
__global__ void pack_b(const float* __restrict__ W, bf16* __restrict__ out,
                       int N, int K) {
  int idx = blockIdx.x * 256 + threadIdx.x;
  if (idx >= N * K) return;
  int e    = idx & 15;
  int lane = (idx >> 4) & 31;
  int frag = idx >> 9;
  int ntn  = N >> 4;
  int nt   = frag % ntn;
  int kt   = frag / ntn;
  int n = nt * 16 + (lane & 15);
  int k = kt * 32 + ((lane >> 4) << 4) + e;
  out[idx] = (bf16)W[(size_t)n * K + k];
}

// ---------------------------------------------------------------------------
// LayerNorm (one wave per row of 192).  shifted=1: fused roll(-1,-3,-3) +
// window-partition gather for the pre-attention path.
// ---------------------------------------------------------------------------
__global__ void __launch_bounds__(256)
ln_kernel(const float* __restrict__ x, const float* __restrict__ g,
          const float* __restrict__ bta, bf16* __restrict__ A, int shifted) {
  int row  = blockIdx.x * 8 + (threadIdx.x >> 5);
  int lane = threadIdx.x & 31;
  const float* src;
  if (shifted) {
    int w = row / NTOK, t = row - w * NTOK;
    int bb = w >> 8, wi = w & 255;
    int wd = wi >> 6, wh = (wi >> 3) & 7, ww = wi & 7;
    int td = t / 49, th = (t / 7) % 7, tw = t % 7;
    int d = wd * 2 + td, h = wh * 7 + th, q = ww * 7 + tw;
    int sd = (d + 1) & 7, sh = (h + 3) % 56, sw = (q + 3) % 56;
    src = x + ((size_t)bb * 25088 + ((size_t)sd * 56 + sh) * 56 + sw) * DIMC;
  } else {
    src = x + (size_t)row * DIMC;
  }
  float v[6], s = 0.f, ss = 0.f;
#pragma unroll
  for (int i = 0; i < 6; ++i) {
    v[i] = src[lane + 32 * i];
    s += v[i];
    ss += v[i] * v[i];
  }
#pragma unroll
  for (int o = 16; o; o >>= 1) {
    s  += __shfl_xor(s, o, 32);
    ss += __shfl_xor(ss, o, 32);
  }
  float mu = s * (1.f / DIMC);
  float rs = rsqrtf(ss * (1.f / DIMC) - mu * mu + 1e-5f);
  bf16* dst = A + (size_t)row * DIMC;
#pragma unroll
  for (int i = 0; i < 6; ++i) {
    int c = lane + 32 * i;
    dst[c] = (bf16)((v[i] - mu) * rs * g[c] + bta[c]);
  }
}

// ---------------------------------------------------------------------------
// QKV GEMM: C[50176,576] = A[50176,192] * Wqkv^T (+bias).  Block tile 128x64,
// 8 waves each computing a 32x32 patch (2x2 WMMA tiles).  The 24KB B tile for
// the block is staged once into LDS via async copies and shared by all waves.
// Epilogue scatters into padded per-(window,head) Q [112,32], K [112,32],
// Vt [32,128] (V^T), with Q pre-scaled by 1/sqrt(hd).
// ---------------------------------------------------------------------------
__global__ void __launch_bounds__(256)
qkv_gemm(const bf16* __restrict__ A, const bf16* __restrict__ Bp,
         const float* __restrict__ bias, bf16* __restrict__ Q,
         bf16* __restrict__ Kb, bf16* __restrict__ Vt) {
  __shared__ bf16 Bs[6 * 4 * 512];  // [kt 0..5][nt_local 0..3] 1KB blobs = 24KB
  int wid = threadIdx.x >> 5, lane = threadIdx.x & 31;
  int wm = wid & 3, wn = wid >> 2;
  int m0 = blockIdx.x * 128 + wm * 32;
  int n0 = blockIdx.y * 64 + wn * 32;

  // Async-stage the block's B fragments: 1536 x 16B chunks, 6 per thread.
  {
    int nt0 = blockIdx.y * 4;
    unsigned lds_base = (unsigned)(uintptr_t)(&Bs[0]);
#pragma unroll
    for (int i = 0; i < 6; ++i) {
      int c = threadIdx.x + 256 * i;       // chunk id, 64 chunks per blob
      int blob = c >> 6;
      int kt = blob >> 2, ntb = blob & 3;
      const bf16* gp = Bp + ((size_t)(kt * 36 + nt0 + ntb) << 9) + ((c & 63) << 3);
      unsigned lds = lds_base + (unsigned)(((blob << 9) + ((c & 63) << 3)) * 2);
      async_copy_b128(lds, (const void*)gp);
    }
  }
  wait_async0();
  __syncthreads();

  f32x8 acc[2][2] = {};
#pragma unroll
  for (int kt = 0; kt < 6; ++kt) {
    bf16x16 a0 = load_afrag(A, m0, DIMC, kt * 32);
    bf16x16 a1 = load_afrag(A, m0 + 16, DIMC, kt * 32);
    bf16x16 b0 = load_bpack_lds(Bs, kt * 4 + wn * 2);
    bf16x16 b1 = load_bpack_lds(Bs, kt * 4 + wn * 2 + 1);
    acc[0][0] = wmma_bf16(a0, b0, acc[0][0]);
    acc[0][1] = wmma_bf16(a0, b1, acc[0][1]);
    acc[1][0] = wmma_bf16(a1, b0, acc[1][0]);
    acc[1][1] = wmma_bf16(a1, b1, acc[1][1]);
  }
  int r = lane >> 4, nl = lane & 15;
  const float scale = 0.17677669529663687f;  // 1/sqrt(32)
#pragma unroll
  for (int tm = 0; tm < 2; ++tm)
#pragma unroll
    for (int tn = 0; tn < 2; ++tn)
#pragma unroll
      for (int j = 0; j < 8; ++j) {
        int m = m0 + tm * 16 + j + 8 * r;
        int n = n0 + tn * 16 + nl;
        float v = acc[tm][tn][j] + bias[n];
        int s = n % 3, dd = (n / 3) & 31, head = n / 96;
        int w = m / NTOK, t = m - w * NTOK;
        size_t wh = (size_t)w * HEADS + head;
        if (s == 0)      Q [(wh * NPAD + t) * HD + dd] = (bf16)(v * scale);
        else if (s == 1) Kb[(wh * NPAD + t) * HD + dd] = (bf16)v;
        else             Vt[(wh * HD + dd) * KPAD + t] = (bf16)v;
      }
}

// Zero the padded token regions so WMMA tiles over the pads are harmless.
__global__ void pad_qkv(bf16* __restrict__ Q, bf16* __restrict__ Kb,
                        bf16* __restrict__ Vt) {
  const int PER = 448 + 448 + 960;  // Q pad + K pad + Vt pad per (w,head)
  int idx = blockIdx.x * 256 + threadIdx.x;
  if (idx >= NWIN * HEADS * PER) return;
  int wh = idx / PER, rem = idx - wh * PER;
  if (rem < 448) {
    int t = NTOK + (rem >> 5), dd = rem & 31;
    Q[((size_t)wh * NPAD + t) * HD + dd] = (bf16)0.f;
  } else if (rem < 896) {
    rem -= 448;
    int t = NTOK + (rem >> 5), dd = rem & 31;
    Kb[((size_t)wh * NPAD + t) * HD + dd] = (bf16)0.f;
  } else {
    rem -= 896;
    int dd = rem / 30, t = NTOK + rem % 30;
    Vt[((size_t)wh * HD + dd) * KPAD + t] = (bf16)0.f;
  }
}

// ---------------------------------------------------------------------------
// Attention: one block per (window, head).  S = Q*K^T (49 WMMA, K=32),
// softmax with in-kernel relative-position bias + shift mask, O = P*V
// (14 tiles x 4 WMMA).  Epilogue: window-reverse + roll(+1,+3,+3) + residual.
// ---------------------------------------------------------------------------
#define LDS_S 132   // f32 score stride (pad -> stride%256B==16B, no bank camp)
#define LDS_P 136   // bf16 P stride
#define ATTN_SMEM (NPAD * LDS_S * 4 + NPAD * LDS_P * 2 + NPAD * 4 + NPAD * 4)

__global__ void __launch_bounds__(256)
attn_kernel(const bf16* __restrict__ Q, const bf16* __restrict__ Kb,
            const bf16* __restrict__ Vt, const float* __restrict__ rel,
            const float* __restrict__ x, float* __restrict__ x2) {
  extern __shared__ char smem[];
  float* S      = (float*)smem;
  bf16*  P      = (bf16*)(smem + NPAD * LDS_S * 4);
  float* rowsum = (float*)(smem + NPAD * LDS_S * 4 + NPAD * LDS_P * 2);
  int*   rid    = (int*)(rowsum + NPAD);

  int w = blockIdx.x, head = blockIdx.y;
  int tid = threadIdx.x, wid = tid >> 5, lane = tid & 31;
  int bb = w >> 8, wi = w & 255;
  int wd = wi >> 6, wh = (wi >> 3) & 7, ww = wi & 7;

  if (tid < NPAD) {
    int t = tid;
    int td = t / 49, th = (t / 7) % 7, tw = t % 7;
    int d = wd * 2 + td, h = wh * 7 + th, q = ww * 7 + tw;
    int rd = d < 6 ? 0 : (d == 6 ? 1 : 2);
    int rh = h < 49 ? 0 : (h < 53 ? 1 : 2);
    int rw = q < 49 ? 0 : (q < 53 ? 1 : 2);
    rid[t] = rd * 9 + rh * 3 + rw;
  }
  size_t qb = ((size_t)w * HEADS + head) * (NPAD * HD);
  const bf16* Qp = Q + qb;
  const bf16* Kp = Kb + qb;
  const bf16* Vp = Vt + ((size_t)w * HEADS + head) * (HD * KPAD);
  __syncthreads();

  // ---- scores ----
  for (int tile = wid; tile < 49; tile += 8) {
    int mt = tile / 7, nt = tile - mt * 7;
    bf16x16 a = load_afrag(Qp, mt * 16, HD, 0);
    bf16x16 b = load_bfrag(Kp, nt * 16, HD, 0);
    f32x8 c = {};
    c = wmma_bf16(a, b, c);
    int r = lane >> 4, nl = lane & 15;
#pragma unroll
    for (int j = 0; j < 8; ++j)
      S[(mt * 16 + j + 8 * r) * LDS_S + nt * 16 + nl] = c[j];
  }
  __syncthreads();

  // ---- bias + mask + softmax (one wave per row) ----
  for (int row = wid; row < NPAD; row += 8) {
    if (row < NTOK) {
      int tdr = row / 49, thr_ = (row / 7) % 7, twr = row % 7;
      float vals[4];
      float mx = -3.0e38f;
#pragma unroll
      for (int ci = 0; ci < 4; ++ci) {
        int c = lane + 32 * ci;
        float v = -3.0e38f;
        if (c < NTOK) {
          int tdc = c / 49, thc = (c / 7) % 7, twc = c % 7;
          int ridx = (tdr - tdc + 1) * 169 + (thr_ - thc + 6) * 13 + (twr - twc + 6);
          v = S[row * LDS_S + c] + rel[ridx * HEADS + head] +
              (rid[row] == rid[c] ? 0.f : -100.f);
        }
        vals[ci] = v;
        mx = fmaxf(mx, v);
      }
#pragma unroll
      for (int o = 16; o; o >>= 1) mx = fmaxf(mx, __shfl_xor(mx, o, 32));
      float sum = 0.f;
#pragma unroll
      for (int ci = 0; ci < 4; ++ci) {
        int c = lane + 32 * ci;
        float e = (c < NTOK) ? __expf(vals[ci] - mx) : 0.f;
        sum += e;
        P[row * LDS_P + c] = (bf16)e;
      }
#pragma unroll
      for (int o = 16; o; o >>= 1) sum += __shfl_xor(sum, o, 32);
      if (lane == 0) rowsum[row] = sum;
    } else {
#pragma unroll
      for (int ci = 0; ci < 4; ++ci) P[row * LDS_P + lane + 32 * ci] = (bf16)0.f;
      if (lane == 0) rowsum[row] = 1.f;
    }
  }
  __syncthreads();

  // ---- O = P * V ----
  for (int tile = wid; tile < 14; tile += 8) {
    int mt = tile % 7, nt = tile / 7;
    f32x8 c = {};
#pragma unroll
    for (int kt = 0; kt < 4; ++kt) {
      bf16x16 a = load_afrag(P, mt * 16, LDS_P, kt * 32);
      bf16x16 b = load_bfrag(Vp, nt * 16, KPAD, kt * 32);
      c = wmma_bf16(a, b, c);
    }
    int r = lane >> 4, nl = lane & 15;
#pragma unroll
    for (int j = 0; j < 8; ++j) {
      int t = mt * 16 + j + 8 * r;
      if (t < NTOK) {
        int n = nt * 16 + nl;
        int td = t / 49, th = (t / 7) % 7, tw = t % 7;
        int d = wd * 2 + td, h = wh * 7 + th, q = ww * 7 + tw;
        int sd = (d + 1) & 7, sh = (h + 3) % 56, sw = (q + 3) % 56;
        size_t off = ((size_t)bb * 25088 + ((size_t)sd * 56 + sh) * 56 + sw) * DIMC +
                     head * HD + n;
        x2[off] = c[j] * (1.f / rowsum[t]) + x[off];
      }
    }
  }
}

// ---------------------------------------------------------------------------
// Fused MLP: hidden tile (128 x 768 bf16 = 198KB) lives in LDS (CDNA5 WGP has
// 320KB).  Phase 1: H = relu(A*W1 + b1); Phase 2: out = relu(H*W2 + b2) + x2.
// ---------------------------------------------------------------------------
#define LDS_H 776  // 768 + 8 pad: row stride 1552B, avoids LDS bank camping
#define MLP_SMEM (128 * LDS_H * 2)

__global__ void __launch_bounds__(256)
mlp_fused(const bf16* __restrict__ A2, const bf16* __restrict__ W1p,
          const float* __restrict__ b1, const bf16* __restrict__ W2p,
          const float* __restrict__ b2, const float* __restrict__ x2,
          float* __restrict__ out) {
  extern __shared__ char smem[];
  bf16* H = (bf16*)smem;
  int wid = threadIdx.x >> 5, lane = threadIdx.x & 31;
  int wm = wid & 3, wn = wid >> 2;
  int m0 = blockIdx.x * 128;
  int r = lane >> 4, nl = lane & 15;

  // Hoist A fragments (reused across all 12 hidden-column chunks).
  bf16x16 a[6][2];
#pragma unroll
  for (int kt = 0; kt < 6; ++kt) {
    a[kt][0] = load_afrag(A2, m0 + wm * 32, DIMC, kt * 32);
    a[kt][1] = load_afrag(A2, m0 + wm * 32 + 16, DIMC, kt * 32);
  }
  for (int nc = 0; nc < 12; ++nc) {
    int n0 = nc * 64 + wn * 32;
    f32x8 acc[2][2] = {};
#pragma unroll
    for (int kt = 0; kt < 6; ++kt) {
      bf16x16 b0 = load_bpack(W1p, kt, n0 >> 4, 48);
      bf16x16 b1v = load_bpack(W1p, kt, (n0 >> 4) + 1, 48);
      acc[0][0] = wmma_bf16(a[kt][0], b0, acc[0][0]);
      acc[0][1] = wmma_bf16(a[kt][0], b1v, acc[0][1]);
      acc[1][0] = wmma_bf16(a[kt][1], b0, acc[1][0]);
      acc[1][1] = wmma_bf16(a[kt][1], b1v, acc[1][1]);
    }
#pragma unroll
    for (int tm = 0; tm < 2; ++tm)
#pragma unroll
      for (int tn = 0; tn < 2; ++tn)
#pragma unroll
        for (int j = 0; j < 8; ++j) {
          int ml = wm * 32 + tm * 16 + j + 8 * r;
          int n = n0 + tn * 16 + nl;
          H[ml * LDS_H + n] = (bf16)fmaxf(acc[tm][tn][j] + b1[n], 0.f);
        }
  }
  __syncthreads();
  for (int oc = 0; oc < 3; ++oc) {
    int n0 = oc * 64 + wn * 32;
    f32x8 acc[2][2] = {};
#pragma unroll
    for (int kt = 0; kt < 24; ++kt) {
      bf16x16 h0 = load_afrag(H, wm * 32, LDS_H, kt * 32);
      bf16x16 h1 = load_afrag(H, wm * 32 + 16, LDS_H, kt * 32);
      bf16x16 b0 = load_bpack(W2p, kt, n0 >> 4, 12);
      bf16x16 b1v = load_bpack(W2p, kt, (n0 >> 4) + 1, 12);
      acc[0][0] = wmma_bf16(h0, b0, acc[0][0]);
      acc[0][1] = wmma_bf16(h0, b1v, acc[0][1]);
      acc[1][0] = wmma_bf16(h1, b0, acc[1][0]);
      acc[1][1] = wmma_bf16(h1, b1v, acc[1][1]);
    }
#pragma unroll
    for (int tm = 0; tm < 2; ++tm)
#pragma unroll
      for (int tn = 0; tn < 2; ++tn)
#pragma unroll
        for (int j = 0; j < 8; ++j) {
          size_t m = (size_t)m0 + wm * 32 + tm * 16 + j + 8 * r;
          int n = n0 + tn * 16 + nl;
          size_t off = m * DIMC + n;
          out[off] = fmaxf(acc[tm][tn][j] + b2[n], 0.f) + x2[off];
        }
  }
}

// ---------------------------------------------------------------------------
extern "C" void kernel_launch(void* const* d_in, const int* in_sizes, int n_in,
                              void* d_out, int out_size, void* d_ws, size_t ws_size,
                              hipStream_t stream) {
  (void)in_sizes; (void)n_in; (void)out_size; (void)ws_size;
  const float* x    = (const float*)d_in[0];
  const float* n1g  = (const float*)d_in[1];
  const float* n1b  = (const float*)d_in[2];
  const float* qkvw = (const float*)d_in[3];
  const float* qkvb = (const float*)d_in[4];
  const float* rel  = (const float*)d_in[5];
  const float* n2g  = (const float*)d_in[6];
  const float* n2b  = (const float*)d_in[7];
  const float* w1   = (const float*)d_in[8];
  const float* b1   = (const float*)d_in[9];
  const float* w2   = (const float*)d_in[10];
  const float* b2   = (const float*)d_in[11];
  float* out = (float*)d_out;

  char* ws = (char*)d_ws;
  size_t off = 0;
  auto alloc = [&](size_t bytes) {
    char* p = ws + off;
    off += (bytes + 255) & ~(size_t)255;
    return p;
  };
  bf16*  A   = (bf16*)alloc((size_t)MROWS * DIMC * 2);
  bf16*  Wqp = (bf16*)alloc((size_t)576 * 192 * 2);
  bf16*  W1p = (bf16*)alloc((size_t)768 * 192 * 2);
  bf16*  W2p = (bf16*)alloc((size_t)192 * 768 * 2);
  bf16*  Qb  = (bf16*)alloc((size_t)NWIN * HEADS * NPAD * HD * 2);
  bf16*  Kb  = (bf16*)alloc((size_t)NWIN * HEADS * NPAD * HD * 2);
  bf16*  Vt  = (bf16*)alloc((size_t)NWIN * HEADS * HD * KPAD * 2);
  float* x2  = (float*)alloc((size_t)MROWS * DIMC * 4);

  pack_b<<<(576 * 192 + 255) / 256, 256, 0, stream>>>(qkvw, Wqp, 576, 192);
  pack_b<<<(768 * 192 + 255) / 256, 256, 0, stream>>>(w1, W1p, 768, 192);
  pack_b<<<(192 * 768 + 255) / 256, 256, 0, stream>>>(w2, W2p, 192, 768);

  ln_kernel<<<MROWS / 8, 256, 0, stream>>>(x, n1g, n1b, A, 1);
  qkv_gemm<<<dim3(MROWS / 128, 576 / 64), 256, 0, stream>>>(A, Wqp, qkvb, Qb, Kb, Vt);
  pad_qkv<<<(NWIN * HEADS * 1856 + 255) / 256, 256, 0, stream>>>(Qb, Kb, Vt);
  attn_kernel<<<dim3(NWIN, HEADS), 256, ATTN_SMEM, stream>>>(Qb, Kb, Vt, rel, x, x2);
  ln_kernel<<<MROWS / 8, 256, 0, stream>>>(x2, n2g, n2b, A, 0);
  mlp_fused<<<MROWS / 128, 256, MLP_SMEM, stream>>>(A, W1p, b1, W2p, b2, x2, out);
}
